// Predictor_14362370638542
// MI455X (gfx1250) — compile-verified, matrix operand order checked by reading, not run on hardware
//
#include <hip/hip_runtime.h>
#include <hip/hip_bf16.h>

// ---------------------------------------------------------------------------
// Problem constants (match reference)
// ---------------------------------------------------------------------------
#define N_NODES    20000
#define N_EDGES    320000
#define TOT_EDGES  (N_EDGES + N_NODES)   // self-loops appended
#define NUM_GRAPHS 128
#define HEADS      8
#define CHAN       64
#define FEAT       512                   // HEADS*CHAN
#define NEG_SLOPE  0.2f

typedef __attribute__((ext_vector_type(16))) __bf16 v16bf;
typedef __attribute__((ext_vector_type(8)))  __bf16 v8bf;
typedef __attribute__((ext_vector_type(8)))  float  v8f;
typedef __attribute__((ext_vector_type(4)))  unsigned int u32x4;
typedef __attribute__((ext_vector_type(8)))  int i32x8;
typedef __attribute__((ext_vector_type(4)))  int i32x4;

#if __has_builtin(__builtin_amdgcn_tensor_load_to_lds)
#define USE_TDM 1
#else
#define USE_TDM 0
#endif

// ---------------------------------------------------------------------------
// helpers
// ---------------------------------------------------------------------------
static __device__ __forceinline__ __bf16 f2bf(float f) {
    union { float f; unsigned u; } in; in.f = f;
    unsigned u = in.u;
    unsigned r = u + 0x7FFFu + ((u >> 16) & 1u);   // round-to-nearest-even
    union { unsigned short s; __bf16 b; } out;
    out.s = (unsigned short)(r >> 16);
    return out.b;
}

static __device__ __forceinline__ float atomicMaxFloat(float* addr, float val) {
    if (val >= 0.0f) {
        return __int_as_float(atomicMax((int*)addr, __float_as_int(val)));
    } else {
        return __uint_as_float(atomicMin((unsigned int*)addr, __float_as_uint(val)));
    }
}

#if USE_TDM
// ---------------------------------------------------------------------------
// Tensor Data Mover: DMA a 128-row x 32-element (bf16) 2D tile into LDS.
// D# per CDNA5 ISA ch.8:
//   group0: [1:0]=count=1, [63:32]=lds_addr, [120:64]=global_addr, [127:126]=2
//   group1: [17:16]=data_size(1->2B), [20]=pad_enable,
//           [24:22]=pad_interval(3 -> 16 DWORDs), [31:25]=pad_amount(3 -> 4 DWORDs)
//           -> LDS row stride = 64B data + 16B pad = 80B (RS=40 elements)
//           [79:48]=tensor_dim0, [111:80]=tensor_dim1 (rows; OOB rows load 0),
//           [127:112]=tile_dim0=32, [143:128]=tile_dim1=128,
//           [207:160]=tensor_dim0_stride (elements)
// ---------------------------------------------------------------------------
static __device__ __forceinline__ void tdm_load_tile(
    unsigned ldsOff, const __bf16* gaddr, int rowsRemain, int strideElems)
{
    unsigned long long ga = (unsigned long long)(size_t)gaddr;
    u32x4 g0 = {};
    g0[0] = 1u;                                            // count=1 (valid user D#)
    g0[1] = ldsOff;                                        // lds_addr (bytes)
    g0[2] = (unsigned)ga;                                  // global_addr[31:0]
    g0[3] = (unsigned)((ga >> 32) & 0x01FFFFFFu) | (2u << 30);  // addr[56:32] | type=2

    i32x8 g1 = {};
    g1[0] = (1 << 16) | (1 << 20) | (3 << 22) | (3 << 25); // data_size|pad cfg
    unsigned td0 = (unsigned)strideElems;                  // row length (>= 32)
    unsigned td1 = (unsigned)rowsRemain;                   // rows remaining (OOB -> 0)
    g1[1] = (int)((td0 & 0xFFFFu) << 16);
    g1[2] = (int)(((td0 >> 16) & 0xFFFFu) | ((td1 & 0xFFFFu) << 16));
    g1[3] = (int)(((td1 >> 16) & 0xFFFFu) | (32u << 16));  // tile_dim0 = 32
    g1[4] = 128;                                           // tile_dim1 = 128
    g1[5] = (int)(unsigned)strideElems;                    // tensor_dim0_stride
    i32x4 g2 = {};
    i32x4 g3 = {};
#if defined(__clang_major__) && (__clang_major__ >= 23)
    i32x8 g4 = {};
    __builtin_amdgcn_tensor_load_to_lds(g0, g1, g2, g3, g4, 0);
#else
    __builtin_amdgcn_tensor_load_to_lds(g0, g1, g2, g3, 0);
#endif
}
#endif

// ---------------------------------------------------------------------------
// f32 -> bf16 conversion kernels (pad K with zeros, weights also transposed)
// ---------------------------------------------------------------------------
__global__ void conv_h_bf16(const float* __restrict__ in, __bf16* __restrict__ out,
                            int M, int Kin, int Kpad)
{
    int idx = blockIdx.x * blockDim.x + threadIdx.x;
    if (idx >= M * Kpad) return;
    int n = idx / Kpad, k = idx - n * Kpad;
    out[idx] = f2bf((k < Kin) ? in[(long)n * Kin + k] : 0.0f);
}

// W [Kin][512] f32 -> Wt [512][Kpad] bf16
__global__ void conv_wt_bf16(const float* __restrict__ W, __bf16* __restrict__ Wt,
                             int Kin, int Kpad)
{
    int idx = blockIdx.x * blockDim.x + threadIdx.x;
    if (idx >= FEAT * Kpad) return;
    int n = idx / Kpad, k = idx - n * Kpad;
    Wt[idx] = f2bf((k < Kin) ? W[(long)k * FEAT + n] : 0.0f);
}

// ---------------------------------------------------------------------------
// WMMA GEMM:  C[M,N] = A[M,K] * Bt[N,K]^T   (A,Bt bf16, K-major, K%32==0)
// block = 256 threads (8 waves), tile 128x128, K-step 32.
// LDS double-buffered, rows stored plain K-major with 80B stride (TDM pad).
// Staging: wave 0 issues TENSOR_LOAD_TO_LDS for A and B tiles of step k+1
// while all 8 waves run the WMMA chain of step k; s_wait_tensorcnt + barrier.
// Fragment read: two ds_load_b128 per matrix at element offsets {lo, lo+16},
// lo = 0 (lanes 0-15: K 0-7 / 16-23) or 8 (lanes 16-31: K 8-15 / 24-31).
// ---------------------------------------------------------------------------
#define RS 40   // LDS row stride in bf16 elements (64B data + 16B pad)

__global__ __launch_bounds__(256)
void gemm_bf16_wmma(const __bf16* __restrict__ A, const __bf16* __restrict__ Bt,
                    float* __restrict__ C, int M, int KA, int N)
{
    __shared__ __bf16 lsA[2][128 * RS];
    __shared__ __bf16 lsB[2][128 * RS];

    const int tid   = threadIdx.x;
    const int wave  = tid >> 5;
    const int lane  = tid & 31;
    const int mBase = blockIdx.y * 128;
    const int nBase = blockIdx.x * 128;

    v8f acc[8] = {};

#if USE_TDM
    const unsigned ldsA0 = (unsigned)(size_t)&lsA[0][0];
    const unsigned ldsB0 = (unsigned)(size_t)&lsB[0][0];
    auto stage = [&](int buf, int k0) {
        if (wave == 0) {
            tdm_load_tile(ldsA0 + buf * (128 * RS * 2),
                          A + (size_t)mBase * KA + k0, M - mBase, KA);
            tdm_load_tile(ldsB0 + buf * (128 * RS * 2),
                          Bt + (size_t)nBase * KA + k0, N - nBase, KA);
        }
    };
    auto stage_wait = [&]() {
        if (wave == 0) __builtin_amdgcn_s_wait_tensorcnt(0);
    };
#else
    // manual fallback: 2 granules (16B) per tile per thread, plain layout
    const int row0 = tid >> 2,         dg0 = (tid & 3) * 8;
    const int row1 = (tid + 256) >> 2, dg1 = ((tid + 256) & 3) * 8;
    auto stage = [&](int buf, int k0) {
        {
            v8bf ga = {};
            int gr = mBase + row0;
            if (gr < M) ga = *(const v8bf*)(A + (size_t)gr * KA + k0 + dg0);
            *(v8bf*)(&lsA[buf][row0 * RS + dg0]) = ga;
            *(v8bf*)(&lsB[buf][row0 * RS + dg0]) =
                *(const v8bf*)(Bt + (size_t)(nBase + row0) * KA + k0 + dg0);
        }
        {
            v8bf ga = {};
            int gr = mBase + row1;
            if (gr < M) ga = *(const v8bf*)(A + (size_t)gr * KA + k0 + dg1);
            *(v8bf*)(&lsA[buf][row1 * RS + dg1]) = ga;
            *(v8bf*)(&lsB[buf][row1 * RS + dg1]) =
                *(const v8bf*)(Bt + (size_t)(nBase + row1) * KA + k0 + dg1);
        }
    };
    auto stage_wait = [&]() {};
#endif

    const int lo      = (lane & 16) >> 1;                    // 0 or 8 elements
    const int arowOff = (wave * 16 + (lane & 15)) * RS + lo;

    stage(0, 0);
    stage_wait();
    __syncthreads();

    int cur = 0;
    for (int k0 = 0; k0 < KA; k0 += 32) {
        // kick off DMA of next K-step into the other LDS buffer
        if (k0 + 32 < KA) stage(cur ^ 1, k0 + 32);

        // A fragment
        v8bf alo = *(const v8bf*)(&lsA[cur][arowOff]);
        v8bf ahi = *(const v8bf*)(&lsA[cur][arowOff + 16]);
        v16bf a = __builtin_shufflevector(alo, ahi,
                  0, 1, 2, 3, 4, 5, 6, 7, 8, 9, 10, 11, 12, 13, 14, 15);

        // all 8 B fragments, then 8 back-to-back WMMAs
        v16bf bfrag[8];
        #pragma unroll
        for (int nt = 0; nt < 8; ++nt) {
            const int boff = (nt * 16 + (lane & 15)) * RS + lo;
            v8bf blo = *(const v8bf*)(&lsB[cur][boff]);
            v8bf bhi = *(const v8bf*)(&lsB[cur][boff + 16]);
            bfrag[nt] = __builtin_shufflevector(blo, bhi,
                        0, 1, 2, 3, 4, 5, 6, 7, 8, 9, 10, 11, 12, 13, 14, 15);
        }
        #pragma unroll
        for (int nt = 0; nt < 8; ++nt) {
            acc[nt] = __builtin_amdgcn_wmma_f32_16x16x32_bf16(
                false, a, false, bfrag[nt], (short)0, acc[nt], false, false);
        }

        if (k0 + 32 < KA) stage_wait();
        __syncthreads();
        cur ^= 1;
    }

    // store, documented C/D layout: VGPR r -> M = r (+8 hi half)
    const int rowOff = wave * 16 + ((lane < 16) ? 0 : 8);
    #pragma unroll
    for (int nt = 0; nt < 8; ++nt) {
        const int col = nBase + nt * 16 + (lane & 15);
        #pragma unroll
        for (int r = 0; r < 8; ++r) {
            const int row = mBase + rowOff + r;
            if (row < M) C[(long)row * N + col] = acc[nt][r];
        }
    }
}

// ---------------------------------------------------------------------------
// fill
// ---------------------------------------------------------------------------
__global__ void fill_f32(float* p, float v, int n) {
    int i = blockIdx.x * blockDim.x + threadIdx.x;
    if (i < n) p[i] = v;
}

// ---------------------------------------------------------------------------
// per-node attention logits: as[n,h] = <h[n,h,:], a_src[h,:]>, same for ad
// ---------------------------------------------------------------------------
__global__ void attn_logits(const float* __restrict__ h,
                            const float* __restrict__ a_src,
                            const float* __restrict__ a_dst,
                            float* __restrict__ as_, float* __restrict__ ad_)
{
    int idx = blockIdx.x * blockDim.x + threadIdx.x;
    if (idx >= N_NODES * HEADS) return;
    int n  = idx >> 3;
    int hd = idx & 7;
    const float* hp = h + (long)n * FEAT + hd * CHAN;
    const float* ws = a_src + hd * CHAN;
    const float* wd = a_dst + hd * CHAN;
    float ss = 0.0f, sd = 0.0f;
    #pragma unroll 8
    for (int c = 0; c < CHAN; ++c) {
        float v = hp[c];
        ss += v * ws[c];
        sd += v * wd[c];
    }
    as_[idx] = ss;
    ad_[idx] = sd;
}

static __device__ __forceinline__ void edge_sd(const long long* ei, int e, int& s, int& d) {
    if (e < N_EDGES) { s = (int)ei[e]; d = (int)ei[N_EDGES + e]; }
    else             { s = e - N_EDGES; d = s; }                   // self-loop
}

// ---------------------------------------------------------------------------
// pass 1: segment max of leaky-relu logits over dst
// ---------------------------------------------------------------------------
__global__ void edge_max(const long long* __restrict__ ei,
                         const float* __restrict__ as_, const float* __restrict__ ad_,
                         float* __restrict__ emax)
{
    int idx = blockIdx.x * blockDim.x + threadIdx.x;
    if (idx >= TOT_EDGES * HEADS) return;
    int e = idx >> 3, hd = idx & 7;
    int s, d; edge_sd(ei, e, s, d);
    float v = as_[s * HEADS + hd] + ad_[d * HEADS + hd];
    v = (v > 0.0f) ? v : NEG_SLOPE * v;
    atomicMaxFloat(&emax[d * HEADS + hd], v);
}

// ---------------------------------------------------------------------------
// pass 2: exp(e - max) per edge, denom accumulation per dst
// ---------------------------------------------------------------------------
__global__ void edge_exp(const long long* __restrict__ ei,
                         const float* __restrict__ as_, const float* __restrict__ ad_,
                         const float* __restrict__ emax,
                         float* __restrict__ den, float* __restrict__ eexp)
{
    int idx = blockIdx.x * blockDim.x + threadIdx.x;
    if (idx >= TOT_EDGES * HEADS) return;
    int e = idx >> 3, hd = idx & 7;
    int s, d; edge_sd(ei, e, s, d);
    float v = as_[s * HEADS + hd] + ad_[d * HEADS + hd];
    v = (v > 0.0f) ? v : NEG_SLOPE * v;
    float ex = __expf(v - emax[d * HEADS + hd]);
    eexp[idx] = ex;
    atomicAdd(&den[d * HEADS + hd], ex);
}

// ---------------------------------------------------------------------------
// pass 3: out[dst,h,:] += (eexp/den[dst,h]) * hlin[src,h,:]
// one wave per (edge, head); 32 lanes cover 64 channels
// ---------------------------------------------------------------------------
__global__ __launch_bounds__(256)
void edge_agg(const long long* __restrict__ ei,
              const float* __restrict__ hlin,
              const float* __restrict__ eexp, const float* __restrict__ den,
              float* __restrict__ agg)
{
    int widx = blockIdx.x * 8 + (threadIdx.x >> 5);
    if (widx >= TOT_EDGES * HEADS) return;
    int lane = threadIdx.x & 31;
    int e = widx >> 3, hd = widx & 7;
    int s, d; edge_sd(ei, e, s, d);
    float alpha = eexp[widx] / den[d * HEADS + hd];
    const float* hp = hlin + (long)s * FEAT + hd * CHAN;
    float*       op = agg  + (long)d * FEAT + hd * CHAN;
    #pragma unroll
    for (int c = lane; c < CHAN; c += 32)
        atomicAdd(&op[c], alpha * hp[c]);
}

// ---------------------------------------------------------------------------
// bias + relu (in-place safe)
// ---------------------------------------------------------------------------
__global__ void bias_relu(const float* __restrict__ agg, const float* __restrict__ b,
                          float* __restrict__ out)
{
    int i = blockIdx.x * blockDim.x + threadIdx.x;
    if (i >= N_NODES * FEAT) return;
    float v = agg[i] + b[i & (FEAT - 1)];
    out[i] = (v > 0.0f) ? v : 0.0f;
}

// ---------------------------------------------------------------------------
// global_add_pool over batch
// ---------------------------------------------------------------------------
__global__ void pool_sum(const float* __restrict__ h, const long long* __restrict__ batch,
                         float* __restrict__ g)
{
    int i = blockIdx.x * blockDim.x + threadIdx.x;
    if (i >= N_NODES * FEAT) return;
    int n = i >> 9, col = i & (FEAT - 1);
    atomicAdd(&g[(int)batch[n] * FEAT + col], h[i]);
}

// ---------------------------------------------------------------------------
// MLP tail
// ---------------------------------------------------------------------------
__global__ __launch_bounds__(64)
void mlp1(const float* __restrict__ g, const float* __restrict__ w,
          const float* __restrict__ b, float* __restrict__ gh)
{
    int graph = blockIdx.x, j = threadIdx.x;
    const float* gp = g + graph * FEAT;
    float s = 0.0f;
    #pragma unroll 8
    for (int k = 0; k < FEAT; ++k) s += gp[k] * w[k * 64 + j];
    s += b[j];
    gh[graph * 64 + j] = (s > 0.0f) ? s : 0.0f;
}

__global__ __launch_bounds__(128)
void mlp2(const float* __restrict__ gh, const float* __restrict__ w,
          const float* __restrict__ b, float* __restrict__ out)
{
    int graph = threadIdx.x;
    if (graph >= NUM_GRAPHS) return;
    const float* gp = gh + graph * 64;
    float s = 0.0f;
    #pragma unroll
    for (int k = 0; k < 64; ++k) s += gp[k] * w[k];
    out[graph] = s + b[0];
}

// ---------------------------------------------------------------------------
// host side
// ---------------------------------------------------------------------------
extern "C" void kernel_launch(void* const* d_in, const int* in_sizes, int n_in,
                              void* d_out, int out_size, void* d_ws, size_t ws_size,
                              hipStream_t stream)
{
    const float* x   = (const float*)d_in[0];
    const float* W[3]  = { (const float*)d_in[1], (const float*)d_in[5], (const float*)d_in[9]  };
    const float* AS[3] = { (const float*)d_in[2], (const float*)d_in[6], (const float*)d_in[10] };
    const float* AD[3] = { (const float*)d_in[3], (const float*)d_in[7], (const float*)d_in[11] };
    const float* BB[3] = { (const float*)d_in[4], (const float*)d_in[8], (const float*)d_in[12] };
    const float* l1w = (const float*)d_in[13];
    const float* l1b = (const float*)d_in[14];
    const float* l2w = (const float*)d_in[15];
    const float* l2b = (const float*)d_in[16];
    const long long* ei    = (const long long*)d_in[17];
    const long long* batch = (const long long*)d_in[18];
    float* out = (float*)d_out;

    size_t off = 0;
    auto carve = [&](size_t bytes) -> void* {
        void* p = (char*)d_ws + off;
        off = (off + bytes + 255) & ~(size_t)255;
        return p;
    };
    const size_t nodeBytes = (size_t)N_NODES * FEAT * sizeof(float);
    float*  lin  = (float*)carve(nodeBytes);
    float*  bufA = (float*)carve(nodeBytes);
    float*  bufB = (float*)carve(nodeBytes);
    __bf16* hb   = (__bf16*)carve((size_t)N_NODES * FEAT * sizeof(__bf16));
    __bf16* wt   = (__bf16*)carve((size_t)FEAT * FEAT * sizeof(__bf16));
    float*  as_  = (float*)carve((size_t)N_NODES * HEADS * sizeof(float));
    float*  ad_  = (float*)carve((size_t)N_NODES * HEADS * sizeof(float));
    float*  emax = (float*)carve((size_t)N_NODES * HEADS * sizeof(float));
    float*  den  = (float*)carve((size_t)N_NODES * HEADS * sizeof(float));
    float*  eexp = (float*)carve((size_t)TOT_EDGES * HEADS * sizeof(float));
    float*  g    = (float*)carve((size_t)NUM_GRAPHS * FEAT * sizeof(float));
    float*  gh   = (float*)carve((size_t)NUM_GRAPHS * 64 * sizeof(float));

    const int nNH = N_NODES * HEADS;
    const int nEH = TOT_EDGES * HEADS;
    const int nNF = N_NODES * FEAT;

    auto run_layer = [&](const float* hin, int Kin, int Kpad, int li, float* agg) {
        conv_h_bf16<<<(N_NODES * Kpad + 255) / 256, 256, 0, stream>>>(hin, hb, N_NODES, Kin, Kpad);
        conv_wt_bf16<<<(FEAT * Kpad + 255) / 256, 256, 0, stream>>>(W[li], wt, Kin, Kpad);
        dim3 ggrid(FEAT / 128, (N_NODES + 127) / 128);
        gemm_bf16_wmma<<<ggrid, 256, 0, stream>>>(hb, wt, lin, N_NODES, Kpad, FEAT);
        attn_logits<<<(nNH + 255) / 256, 256, 0, stream>>>(lin, AS[li], AD[li], as_, ad_);
        fill_f32<<<(nNH + 255) / 256, 256, 0, stream>>>(emax, -3.0e38f, nNH);
        fill_f32<<<(nNH + 255) / 256, 256, 0, stream>>>(den, 0.0f, nNH);
        fill_f32<<<(nNF + 255) / 256, 256, 0, stream>>>(agg, 0.0f, nNF);
        edge_max<<<(nEH + 255) / 256, 256, 0, stream>>>(ei, as_, ad_, emax);
        edge_exp<<<(nEH + 255) / 256, 256, 0, stream>>>(ei, as_, ad_, emax, den, eexp);
        edge_agg<<<(nEH + 7) / 8, 256, 0, stream>>>(ei, lin, eexp, den, agg);
        bias_relu<<<(nNF + 255) / 256, 256, 0, stream>>>(agg, BB[li], agg);
    };

    run_layer(x,    79,  96,  0, bufA);
    run_layer(bufA, 512, 512, 1, bufB);
    run_layer(bufB, 512, 512, 2, bufA);

    fill_f32<<<(NUM_GRAPHS * FEAT + 255) / 256, 256, 0, stream>>>(g, 0.0f, NUM_GRAPHS * FEAT);
    pool_sum<<<(nNF + 255) / 256, 256, 0, stream>>>(bufA, batch, g);
    mlp1<<<NUM_GRAPHS, 64, 0, stream>>>(g, l1w, l1b, gh);
    mlp2<<<1, 128, 0, stream>>>(gh, l2w, l2b, out);

    (void)in_sizes; (void)n_in; (void)out_size; (void)ws_size;
}
